// SwinTransformerBlock_kv_42674795053149
// MI455X (gfx1250) — matrix operand order repaired and checked by reading, not compiled, not run on hardware
//
#include <hip/hip_runtime.h>
#include <hip/hip_bf16.h>

typedef __attribute__((ext_vector_type(16))) _Float16 v16h;
typedef __attribute__((ext_vector_type(8)))  _Float16 v8h;
typedef __attribute__((ext_vector_type(8)))  float    v8f;

#define DIMC   192
#define HEADS  6
#define HDIM   32
#define LTOT   131072
#define HIDDEN 768
#define SCALE  0.17677669529663687f   // 1/sqrt(32)

// ---- WMMA fragment loaders (CDNA5 16x16x32 f16 layouts) ----
// A (16x32, f16): lane r=lane&15 holds row r; halves 0..7 = K (lane>>4)*8..+7,
// halves 8..15 = K 16+(lane>>4)*8..+7  -> two ds/global b128 loads from row-major.
__device__ __forceinline__ v16h ldA(const _Float16* base, int r0, int c0, int stride) {
  int lane = threadIdx.x & 31;
  const _Float16* p = base + (size_t)(r0 + (lane & 15)) * stride + c0 + ((lane >> 4) << 3);
  v8h lo = *(const v8h*)p;
  v8h hi = *(const v8h*)(p + 16);
  v16h a;
#pragma unroll
  for (int i = 0; i < 8; ++i) { a[i] = lo[i]; a[8 + i] = hi[i]; }
  return a;
}
// B (32x16, f16): lane holds col n=lane&15; halves 0..7 = K (lane>>4)*8..+7,
// halves 8..15 = K 16+(lane>>4)*8..+7 of row n of a transposed (NxK) matrix.
__device__ __forceinline__ v16h ldB(const _Float16* base, int n0, int k0, int stride) {
  int lane = threadIdx.x & 31;
  const _Float16* p = base + (size_t)(n0 + (lane & 15)) * stride + k0 + ((lane >> 4) << 3);
  v8h lo = *(const v8h*)p;
  v8h hi = *(const v8h*)(p + 16);
  v16h b;
#pragma unroll
  for (int i = 0; i < 8; ++i) { b[i] = lo[i]; b[8 + i] = hi[i]; }
  return b;
}
#define WMMA_F16(a, b, c) \
  __builtin_amdgcn_wmma_f32_16x16x32_f16(false, (a), false, (b), (short)0, (c), false, false)

// ---- one-time prep: f16-transpose weights, materialize per-head bias ----
__global__ void __launch_bounds__(256) prep_kernel(
    const float* __restrict__ Wkv, const float* __restrict__ Wproj,
    const float* __restrict__ W1,  const float* __restrict__ W2,
    const float* __restrict__ btab, const int* __restrict__ ridx,
    _Float16* __restrict__ WkvT, _Float16* __restrict__ WprojT,
    _Float16* __restrict__ W1T,  _Float16* __restrict__ W2T,
    float* __restrict__ bias6)
{
  int i = blockIdx.x * 256 + threadIdx.x;
  if (i < 73728) { int n = i / 192, k = i % 192; WkvT[i] = (_Float16)Wkv[k * 384 + n]; return; }
  i -= 73728;
  if (i < 36864) { int n = i / 192, k = i % 192; WprojT[i] = (_Float16)Wproj[k * 192 + n]; return; }
  i -= 36864;
  if (i < 147456) { int n = i / 192, k = i % 192; W1T[i] = (_Float16)W1[k * 768 + n]; return; }
  i -= 147456;
  if (i < 147456) { int n = i / 768, k = i % 768; W2T[i] = (_Float16)W2[k * 192 + n]; return; }
  i -= 147456;
  if (i < 24576) { int h = i / 4096; int qk = i % 4096; bias6[i] = btab[ridx[qk] * 6 + h]; }
}

// ---- attention: one block per 4x4x4 window ----
__global__ void __launch_bounds__(256) attn_kernel(
    const float* __restrict__ skip, const float* __restrict__ xup,
    const float* __restrict__ g1, const float* __restrict__ be1,
    const _Float16* __restrict__ WkvT, const float* __restrict__ bkv,
    const _Float16* __restrict__ WprojT, const float* __restrict__ bproj,
    const float* __restrict__ bias6, float* __restrict__ out)
{
  extern __shared__ char smem[];
  _Float16* sA    = (_Float16*)smem;       // 64x192 LN(skip); reused as O (64x192)
  _Float16* uA    = sA + 64 * DIMC;        // 64x192 LN(x_up) = Q source
  _Float16* kS    = uA + 64 * DIMC;        // 64x192 K, row-major (tok, ch)
  _Float16* vT    = kS + 64 * DIMC;        // 192x64 V transposed (ch, tok)
  _Float16* attnA = vT + 64 * DIMC;        // 64x64 probs (per head, reused)
  float*    logit = (float*)(attnA + 64 * 64); // 64x64 f32

  const int w = blockIdx.x;
  const int bb = w >> 11;
  const int wrem = w & 2047;
  const int zb = wrem >> 8, yb = (wrem >> 4) & 15, xb = wrem & 15;
  const int wid = threadIdx.x >> 5, lane = threadIdx.x & 31;
  const int lane15 = lane & 15, half8 = (lane >> 4) << 3;
  const int mrow = wid >> 1, nhalf = wid & 1;   // register-blocked GEMM assignment

  // LayerNorm (norm1) of both streams, gathered through the window mapping
  for (int tt = 0; tt < 8; ++tt) {
    int t = wid * 8 + tt;
    int z = zb * 4 + (t >> 4), y = yb * 4 + ((t >> 2) & 3), x = xb * 4 + (t & 3);
    size_t row = ((size_t)bb * LTOT + ((z << 12) | (y << 6) | x)) * DIMC;
    float vs[6], vu[6];
    float s1 = 0.f, s2 = 0.f, u1 = 0.f, u2 = 0.f;
#pragma unroll
    for (int j = 0; j < 6; ++j) {
      float a = skip[row + lane + 32 * j]; vs[j] = a; s1 += a; s2 += a * a;
      float c = xup [row + lane + 32 * j]; vu[j] = c; u1 += c; u2 += c * c;
    }
#pragma unroll
    for (int o = 16; o > 0; o >>= 1) {
      s1 += __shfl_xor(s1, o); s2 += __shfl_xor(s2, o);
      u1 += __shfl_xor(u1, o); u2 += __shfl_xor(u2, o);
    }
    float ms = s1 * (1.f / 192.f), mu_ = u1 * (1.f / 192.f);
    float is = rsqrtf(s2 * (1.f / 192.f) - ms * ms + 1e-5f);
    float iu = rsqrtf(u2 * (1.f / 192.f) - mu_ * mu_ + 1e-5f);
#pragma unroll
    for (int j = 0; j < 6; ++j) {
      int c = lane + 32 * j;
      float gg = g1[c], bv = be1[c];
      sA[t * DIMC + c] = (_Float16)((vs[j] - ms) * is * gg + bv);
      uA[t * DIMC + c] = (_Float16)((vu[j] - mu_) * iu * gg + bv);
    }
  }
  __syncthreads();

  // KV = LN(skip) @ Wkv + bkv : 64x384. Wave = (m, 12 n-tiles), A reused x12.
  {
    v8f acc[12];
#pragma unroll
    for (int j = 0; j < 12; ++j) {
      float bv = bkv[(nhalf * 12 + j) * 16 + lane15];
#pragma unroll
      for (int r = 0; r < 8; ++r) acc[j][r] = bv;
    }
    for (int kk = 0; kk < 6; ++kk) {
      v16h a = ldA(sA, mrow * 16, kk * 32, DIMC);
#pragma unroll
      for (int j = 0; j < 12; ++j) {
        v16h bfr = ldB(WkvT, (nhalf * 12 + j) * 16, kk * 32, DIMC);
        acc[j] = WMMA_F16(a, bfr, acc[j]);
      }
    }
    if (nhalf == 0) {                   // K half -> row-major (tok, ch)
      int r0 = mrow * 16 + half8;
#pragma unroll
      for (int j = 0; j < 12; ++j) {
        int col = j * 16 + lane15;
#pragma unroll
        for (int r = 0; r < 8; ++r) kS[(r0 + r) * DIMC + col] = (_Float16)acc[j][r];
      }
    } else {                            // V half -> transposed, contiguous b128 store
#pragma unroll
      for (int j = 0; j < 12; ++j) {
        int d = j * 16 + lane15;
        v8h hv;
#pragma unroll
        for (int r = 0; r < 8; ++r) hv[r] = (_Float16)acc[j][r];
        *(v8h*)(vT + d * 64 + mrow * 16 + half8) = hv;
      }
    }
  }
  __syncthreads();

  for (int h = 0; h < HEADS; ++h) {
    // logits = Q_h @ K_h^T : wave = (m, 2 n-tiles), A reused x2, single K=32 step
    {
      const int nb = nhalf * 2;
      v16h a = ldA(uA, mrow * 16, h * HDIM, DIMC);
      v8f acc[2];
#pragma unroll
      for (int j = 0; j < 2; ++j) {
#pragma unroll
        for (int r = 0; r < 8; ++r) acc[j][r] = 0.f;
        v16h bfr = ldB(kS, (nb + j) * 16, h * HDIM, DIMC);
        acc[j] = WMMA_F16(a, bfr, acc[j]);
      }
      int r0 = mrow * 16 + half8;
#pragma unroll
      for (int j = 0; j < 2; ++j) {
        int col = (nb + j) * 16 + lane15;
#pragma unroll
        for (int r = 0; r < 8; ++r) logit[(r0 + r) * 64 + col] = acc[j][r];
      }
    }
    __syncthreads();
    // softmax with bias: 4 lanes per row, shfl_xor reductions
    {
      int row = threadIdx.x >> 2, part = threadIdx.x & 3;
      const float* lp = logit + row * 64 + part * 16;
      const float* bp = bias6 + h * 4096 + row * 64 + part * 16;
      float vals[16], mx = -3.0e38f;
#pragma unroll
      for (int j = 0; j < 16; ++j) { float v = lp[j] * SCALE + bp[j]; vals[j] = v; mx = fmaxf(mx, v); }
      mx = fmaxf(mx, __shfl_xor(mx, 1));
      mx = fmaxf(mx, __shfl_xor(mx, 2));
      float s = 0.f;
#pragma unroll
      for (int j = 0; j < 16; ++j) { float e = __expf(vals[j] - mx); vals[j] = e; s += e; }
      s += __shfl_xor(s, 1);
      s += __shfl_xor(s, 2);
      float inv = 1.f / s;
      _Float16* ap = attnA + row * 64 + part * 16;
#pragma unroll
      for (int j = 0; j < 16; ++j) ap[j] = (_Float16)(vals[j] * inv);
    }
    __syncthreads();
    // O_h = probs @ V_h : 8 tiles, 1 per wave, K=2 steps
    {
      v8f c; for (int r = 0; r < 8; ++r) c[r] = 0.f;
#pragma unroll
      for (int kk = 0; kk < 2; ++kk) {
        v16h a   = ldA(attnA, mrow * 16, kk * 32, 64);
        v16h bfr = ldB(vT + h * HDIM * 64, nhalf * 16, kk * 32, 64);
        c = WMMA_F16(a, bfr, c);
      }
      int col = h * HDIM + nhalf * 16 + lane15, r0 = mrow * 16 + half8;
#pragma unroll
      for (int r = 0; r < 8; ++r) sA[(r0 + r) * DIMC + col] = (_Float16)c[r]; // O aliases sA
    }
  }
  __syncthreads();

  // x = skip + win_rev(O @ Wproj + bproj). Wave = (m, 6 n-tiles), A reused x6.
  {
    v8f acc[6];
#pragma unroll
    for (int j = 0; j < 6; ++j) {
      float bv = bproj[(nhalf * 6 + j) * 16 + lane15];
#pragma unroll
      for (int r = 0; r < 8; ++r) acc[j][r] = bv;
    }
    for (int kk = 0; kk < 6; ++kk) {
      v16h a = ldA(sA, mrow * 16, kk * 32, DIMC);
#pragma unroll
      for (int j = 0; j < 6; ++j) {
        v16h bfr = ldB(WprojT, (nhalf * 6 + j) * 16, kk * 32, DIMC);
        acc[j] = WMMA_F16(a, bfr, acc[j]);
      }
    }
    int r0 = mrow * 16 + half8;
#pragma unroll
    for (int r = 0; r < 8; ++r) {
      int tkn = r0 + r;
      int z = zb * 4 + (tkn >> 4), y = yb * 4 + ((tkn >> 2) & 3), x = xb * 4 + (tkn & 3);
      size_t rowoff = ((size_t)bb * LTOT + ((z << 12) | (y << 6) | x)) * DIMC;
#pragma unroll
      for (int j = 0; j < 6; ++j) {
        size_t idx = rowoff + (nhalf * 6 + j) * 16 + lane15;
        out[idx] = skip[idx] + acc[j][r];
      }
    }
  }
}

// ---- MLP: one block per 64 tokens (token-major, reads/writes d_out in place) ----
__global__ void __launch_bounds__(256) mlp_kernel(
    float* __restrict__ out, const float* __restrict__ g2, const float* __restrict__ be2,
    const _Float16* __restrict__ W1T, const float* __restrict__ b1,
    const _Float16* __restrict__ W2T, const float* __restrict__ b2)
{
  extern __shared__ char smem[];
  _Float16* xln = (_Float16*)smem;      // 64x192
  _Float16* hb  = xln + 64 * DIMC;      // 64x768

  const size_t base = (size_t)blockIdx.x * 64;
  const int wid = threadIdx.x >> 5, lane = threadIdx.x & 31;
  const int lane15 = lane & 15, half8 = (lane >> 4) << 3;
  const int mrow = wid >> 1, nhalf = wid & 1;

  for (int tt = 0; tt < 8; ++tt) {
    int t = wid * 8 + tt;
    const float* row = out + (base + t) * DIMC;
    float v[6]; float s1 = 0.f, s2 = 0.f;
#pragma unroll
    for (int j = 0; j < 6; ++j) { float a = row[lane + 32 * j]; v[j] = a; s1 += a; s2 += a * a; }
#pragma unroll
    for (int o = 16; o > 0; o >>= 1) { s1 += __shfl_xor(s1, o); s2 += __shfl_xor(s2, o); }
    float mu = s1 * (1.f / 192.f);
    float inv = rsqrtf(s2 * (1.f / 192.f) - mu * mu + 1e-5f);
#pragma unroll
    for (int j = 0; j < 6; ++j) {
      int c = lane + 32 * j;
      xln[t * DIMC + c] = (_Float16)((v[j] - mu) * inv * g2[c] + be2[c]);
    }
  }
  __syncthreads();

  // h = gelu(LN(x) @ W1 + b1) : wave = (m, 24 n-tiles) in 3 chunks of 8, A reused x8
  for (int c3 = 0; c3 < 3; ++c3) {
    const int nb = nhalf * 24 + c3 * 8;
    v8f acc[8];
#pragma unroll
    for (int j = 0; j < 8; ++j) {
      float bv = b1[(nb + j) * 16 + lane15];
#pragma unroll
      for (int r = 0; r < 8; ++r) acc[j][r] = bv;
    }
    for (int kk = 0; kk < 6; ++kk) {
      v16h a = ldA(xln, mrow * 16, kk * 32, DIMC);
#pragma unroll
      for (int j = 0; j < 8; ++j) {
        v16h bfr = ldB(W1T, (nb + j) * 16, kk * 32, DIMC);
        acc[j] = WMMA_F16(a, bfr, acc[j]);
      }
    }
    int r0 = mrow * 16 + half8;
#pragma unroll
    for (int j = 0; j < 8; ++j) {
      int col = (nb + j) * 16 + lane15;
#pragma unroll
      for (int r = 0; r < 8; ++r) {
        float v = acc[j][r];
        float g = 0.5f * v * (1.f + erff(v * 0.70710678118654752f));
        hb[(r0 + r) * HIDDEN + col] = (_Float16)g;
      }
    }
  }
  __syncthreads();

  // out = x + h @ W2 + b2 : wave = (m, 6 n-tiles), K=24, A reused x6
  {
    v8f acc[6];
#pragma unroll
    for (int j = 0; j < 6; ++j) {
      float bv = b2[(nhalf * 6 + j) * 16 + lane15];
#pragma unroll
      for (int r = 0; r < 8; ++r) acc[j][r] = bv;
    }
    for (int kk = 0; kk < 24; ++kk) {
      v16h a = ldA(hb, mrow * 16, kk * 32, HIDDEN);
#pragma unroll
      for (int j = 0; j < 6; ++j) {
        v16h bfr = ldB(W2T, (nhalf * 6 + j) * 16, kk * 32, HIDDEN);
        acc[j] = WMMA_F16(a, bfr, acc[j]);
      }
    }
    int r0 = mrow * 16 + half8;
#pragma unroll
    for (int r = 0; r < 8; ++r) {
      size_t rowoff = (base + r0 + r) * DIMC;
#pragma unroll
      for (int j = 0; j < 6; ++j) {
        size_t idx = rowoff + (nhalf * 6 + j) * 16 + lane15;
        out[idx] = out[idx] + acc[j][r];
      }
    }
  }
}

extern "C" void kernel_launch(void* const* d_in, const int* in_sizes, int n_in,
                              void* d_out, int out_size, void* d_ws, size_t ws_size,
                              hipStream_t stream) {
  (void)in_sizes; (void)n_in; (void)out_size; (void)ws_size;
  const float* skip  = (const float*)d_in[0];
  const float* xup   = (const float*)d_in[1];
  const float* g1    = (const float*)d_in[2];
  const float* be1   = (const float*)d_in[3];
  const float* g2    = (const float*)d_in[4];
  const float* be2   = (const float*)d_in[5];
  const float* Wkv   = (const float*)d_in[6];
  const float* bkv   = (const float*)d_in[7];
  const float* Wproj = (const float*)d_in[8];
  const float* bproj = (const float*)d_in[9];
  const float* btab  = (const float*)d_in[10];
  const int*   ridx  = (const int*)d_in[11];
  const float* W1    = (const float*)d_in[12];
  const float* b1    = (const float*)d_in[13];
  const float* W2    = (const float*)d_in[14];
  const float* b2    = (const float*)d_in[15];
  float* out = (float*)d_out;

  char* ws = (char*)d_ws;
  _Float16* WkvT   = (_Float16*)(ws);            // 384x192 f16 = 147456 B
  _Float16* WprojT = (_Float16*)(ws + 147456);   // 192x192 f16 =  73728 B
  _Float16* W1T    = (_Float16*)(ws + 221184);   // 768x192 f16 = 294912 B
  _Float16* W2T    = (_Float16*)(ws + 516096);   // 192x768 f16 = 294912 B
  float*    bias6  = (float*)   (ws + 811008);   // 6x64x64 f32 =  98304 B

  prep_kernel<<<1680, 256, 0, stream>>>(Wkv, Wproj, W1, W2, btab, ridx,
                                        WkvT, WprojT, W1T, W2T, bias6);
  attn_kernel<<<4096, 256, 122880, stream>>>(skip, xup, g1, be1,
                                             WkvT, bkv, WprojT, bproj, bias6, out);
  mlp_kernel<<<4096, 256, 122880, stream>>>(out, g2, be2, W1T, b1, W2T, b2);
}
